// InterpolationEmbeddingModel_17789754540437
// MI455X (gfx1250) — compile-verified, matrix operand order checked by reading, not run on hardware
//
#include <hip/hip_runtime.h>

// ---------------------------------------------------------------------------
// InterpolationEmbedding-LSTM for MI455X (gfx1250, wave32, WMMA 16x16x32 bf16)
//
// B=128, S=1024, IN=16, H=256, 4H=1024, E=32.
// 8 persistent workgroups x 512 threads (16 waves). Each WG owns 16 batch rows
// (WMMA M) and runs the full S=1024 recurrence. Per step each wave owns 4
// N-tiles (64 gate columns): 2 x 9 x 4 = 72 WMMAs + 1 alpha WMMA.
// Weights are pre-swizzled bf16 fragments in d_ws (L2-resident, 1.125 MB).
// v7: fused A/emb LDS arena (one ds base register + immediate offsets) and
//     32-bit activation/output offsets (saddr+voffset global form, no 64-bit
//     VGPR address pairs) -> eliminate the last loop-invariant spills.
// ---------------------------------------------------------------------------

typedef __bf16 bf16_t;
typedef __attribute__((ext_vector_type(16))) __bf16 v16bf;
typedef __attribute__((ext_vector_type(8)))  float  v8f;

#define B_   128
#define S_   1024
#define IN_  16
#define H_   256
#define G_   1024      // 4*H
#define E_   32
#define KT   9         // K tiles of 32: 256 (h) + 16 (x) + 1 (bias) + 15 (zero)
#define NT   64        // N tiles of 16 over G_
#define ROWS 16        // batch rows per workgroup
#define THREADS 512    // 16 waves

#define WFRAG_ELEMS  (2*KT*NT*512)   // 589824 bf16 = 1.125 MB
#define WEFRAG_ELEMS (16*512)        // 8192 bf16

__device__ __forceinline__ float fast_sig(float x) {
  return __builtin_amdgcn_rcpf(1.0f + __expf(-x));
}
__device__ __forceinline__ float fast_tanh(float x) {
#if __has_builtin(__builtin_amdgcn_tanhf)
  return __builtin_amdgcn_tanhf(x);     // gfx1250 v_tanh_f32
#else
  return 2.0f * __builtin_amdgcn_rcpf(1.0f + __expf(-2.0f * x)) - 1.0f;
#endif
}

// K index inside a 32-wide tile for fragment (lane, elem) — ISA 7.12.2 16-bit layout
__device__ __forceinline__ int fragK(int fl, int fe) {
  return (fe < 8 ? fe : fe + 8) + 8 * (fl >> 4);
}

// ---------------------------------------------------------------------------
// Prep: convert augmented [W_h; W_x; b; 0] (K=288) and W_e to bf16 fragments.
// ---------------------------------------------------------------------------
__global__ __launch_bounds__(256)
void prep_weights(const float* __restrict__ Wx1, const float* __restrict__ Wx2,
                  const float* __restrict__ Wh1, const float* __restrict__ Wh2,
                  const float* __restrict__ b1,  const float* __restrict__ b2,
                  const float* __restrict__ We,
                  bf16_t* __restrict__ wFrags, bf16_t* __restrict__ weFrags)
{
  int idx = blockIdx.x * 256 + threadIdx.x;
  if (idx < WFRAG_ELEMS) {
    int wsel = idx / (KT*NT*512);
    int r0   = idx % (KT*NT*512);
    int kt   = r0 / (NT*512);
    int r1   = r0 % (NT*512);
    int nt   = r1 / 512;
    int w2   = r1 % 512;
    int fl   = w2 >> 4, fe = w2 & 15;
    int row  = kt*32 + fragK(fl, fe);
    int n    = nt*16 + (fl & 15);
    const float* Wh = wsel ? Wh2 : Wh1;
    const float* Wx = wsel ? Wx2 : Wx1;
    const float* bb = wsel ? b2  : b1;
    float v = 0.0f;
    if      (row < H_)        v = Wh[(size_t)row*G_ + n];
    else if (row < H_ + IN_)  v = Wx[(size_t)(row - H_)*G_ + n];
    else if (row == H_ + IN_) v = bb[n];
    wFrags[idx] = (bf16_t)v;
  } else if (idx < WFRAG_ELEMS + WEFRAG_ELEMS) {
    int j  = idx - WFRAG_ELEMS;
    int nt = j / 512, w2 = j % 512;
    int fl = w2 >> 4, fe = w2 & 15;
    int K  = fragK(fl, fe);
    int n  = nt*16 + (fl & 15);
    weFrags[j] = (bf16_t)We[(size_t)K*H_ + n];
  }
}

// ---------------------------------------------------------------------------
// Persistent recurrent kernel: one WG = 16 batch rows, all S steps.
// ---------------------------------------------------------------------------
__global__ __launch_bounds__(THREADS)
__attribute__((amdgpu_waves_per_eu(4, 4)))
void lstm_persistent(const int*   __restrict__ eIdx,
                     const float* __restrict__ x,
                     const float* __restrict__ embT,
                     const float* __restrict__ b_e,
                     const float* __restrict__ linW,
                     const float* __restrict__ linB,
                     const bf16_t* __restrict__ wFrags,
                     const bf16_t* __restrict__ weFrags,
                     float* __restrict__ out)
{
  // Fused A-fragment arena: [0 .. KT*512) = GEMM A frags (kt 8 = x-aug),
  // [KT*512 .. (KT+1)*512) = emb A frag. One ds base register per thread,
  // slot selection via the 16-bit ds immediate offset.
  __shared__ bf16_t sAB[(KT+1)*512];
  __shared__ bf16_t sWe[16*512];       // W_e B frags (resident)
  __shared__ float  sAlpha[ROWS*H_];   // 16x256 interpolation gate
  __shared__ float  sG[ROWS*G_];       // pass1: raw g1; pass2: mixed gates
  __shared__ float  sLinW[H_];

  const int tid   = threadIdx.x;
  const int wave  = tid >> 5;          // 0..15
  const int lane  = tid & 31;
  const int l16   = lane & 15;
  const int lg    = lane >> 4;
  const int bBase = blockIdx.x * ROWS;

  for (int i = tid; i < (KT+1)*512; i += THREADS) sAB[i] = (bf16_t)0.0f; // h_0 = 0
  for (int i = tid; i < 16*512; i += THREADS) sWe[i] = weFrags[i];
  if (tid < H_) sLinW[tid] = linW[tid];
  const float lb = linB[0];

  // ---- Hoisted per-thread phase-0 metadata (1 fragment entry per thread) --
  // fl0*16+fe0 == tid, so both store slots hang off ONE base pointer.
  const int fM0 = (tid >> 4) & 15;
  const int fK0 = fragK(tid >> 4, tid & 15);
  const bool isX   = (fK0 < IN_);
  const float cnst = (fK0 == IN_) ? 1.0f : 0.0f;     // aug rows: bias / zero
  bf16_t* const pAB = &sAB[tid];
  // 32-bit offsets -> saddr+voffset global loads, no 64-bit VGPR addr pairs
  const unsigned xoff = (unsigned)(bBase + fM0)*S_*IN_ + (isX ? fK0 : IN_ - 1);
  const unsigned eoff = (unsigned)(bBase + fM0)*S_;

  // alpha bias is constant across t: this wave owns N-tile nt == wave
  const float beV = b_e[wave*16 + l16];

  // cell state: wave owns row uRow == wave; lane owns cols uJ0..uJ0+7
  const int uRow = wave;
  const int uJ0  = lane * 8;
  const unsigned outOff = (unsigned)(bBase + uRow)*S_;
  float c[8];
  #pragma unroll
  for (int i = 0; i < 8; ++i) c[i] = 0.0f;

  const v16bf* wF = (const v16bf*)wFrags;
  // this wave's weight-fragment base: N-tiles [wave*4, wave*4+4)
  const v16bf* wBase = wF + (size_t)(wave*4)*32 + lane;
  const v8f vzero = {0.f,0.f,0.f,0.f,0.f,0.f,0.f,0.f};

  // ---- prologue: two-deep activation pipeline ----------------------------
  float xvN, embN;
  int   eiCur;
  {
    int ei0 = eIdx[eoff];
    embN   = embT[(unsigned)(ei0*E_ + fK0)];
    xvN    = x[xoff];
    eiCur  = eIdx[eoff + 1u];
  }

  for (int t = 0; t < S_; ++t) {
    // -------- Phase 0: commit prefetched x/emb values to A fragments -------
    pAB[8*512]  = (bf16_t)(isX ? xvN : cnst);   // ds offset 8192 B
    pAB[KT*512] = (bf16_t)embN;                 // ds offset 9216 B
    __syncthreads();

    // -------- Phase 1: alpha = sigmoid(emb @ W_e + b_e), 1 tile/wave -------
    {
      v16bf ae = *(const v16bf*)&sAB[KT*512 + lane*16];
      v16bf bw = *(const v16bf*)&sWe[wave*512 + lane*16];
      v8f d = __builtin_amdgcn_wmma_f32_16x16x32_bf16(
                  false, ae, false, bw, (short)0, vzero, false, false);
      #pragma unroll
      for (int r = 0; r < 8; ++r) {
        int row = r + 8*lg;
        sAlpha[row*H_ + wave*16 + l16] = fast_sig(d[r] + beV);  // scalar bias
      }
    }
    __syncthreads();

    // -------- Phase 2a: next step's loads (independent), then g1 GEMM ------
    if (t + 1 < S_) {       // latency hidden under the WMMAs below
      xvN  = x[xoff + (unsigned)(t + 1)*IN_];
      embN = embT[(unsigned)(eiCur*E_ + fK0)];      // index from last step
      unsigned t2 = (t + 2 < S_) ? (unsigned)(t + 2) : (unsigned)(S_ - 1);
      eiCur = eIdx[eoff + t2];
    }

    v8f acc[4];
    {   // kt = 0 peeled: inline-0 C operand starts each chain
      v16bf a = *(const v16bf*)&sAB[lane*16];
      __builtin_prefetch(wBase + (size_t)1*NT*32, 0, 3);
      #pragma unroll
      for (int n = 0; n < 4; ++n) {
        v16bf b = wBase[(size_t)n*32];
        acc[n] = __builtin_amdgcn_wmma_f32_16x16x32_bf16(
                     false, a, false, b, (short)0, vzero, false, false);
      }
    }
    #pragma unroll
    for (int kt = 1; kt < KT; ++kt) {
      v16bf a = *(const v16bf*)&sAB[kt*512 + lane*16];
      if (kt + 1 < KT)   // WGP-scope: pull next W1 K-slab into near caches
        __builtin_prefetch(wBase + (size_t)(kt + 1)*NT*32, 0, 3);
      #pragma unroll
      for (int n = 0; n < 4; ++n) {
        v16bf b = wBase[((size_t)kt*NT + n)*32];
        acc[n] = __builtin_amdgcn_wmma_f32_16x16x32_bf16(
                     false, a, false, b, (short)0, acc[n], false, false);
      }
    }
    // park raw g1 in thread-private sG slots (same-wave LDS ops are in-order)
    #pragma unroll
    for (int n = 0; n < 4; ++n) {
      int col = (wave*4 + n)*16 + l16;
      #pragma unroll
      for (int r = 0; r < 8; ++r)
        sG[(r + 8*lg)*G_ + col] = acc[n][r];
    }

    // -------- Phase 2b: g2 GEMM, then mix gates = g2 + a*(g1-g2) -----------
    {   // kt = 0 peeled
      v16bf a = *(const v16bf*)&sAB[lane*16];
      __builtin_prefetch(wBase + (size_t)(KT + 1)*NT*32, 0, 3);
      #pragma unroll
      for (int n = 0; n < 4; ++n) {
        v16bf b = wBase[((size_t)KT*NT + n)*32];
        acc[n] = __builtin_amdgcn_wmma_f32_16x16x32_bf16(
                     false, a, false, b, (short)0, vzero, false, false);
      }
    }
    #pragma unroll
    for (int kt = 1; kt < KT; ++kt) {
      v16bf a = *(const v16bf*)&sAB[kt*512 + lane*16];
      if (kt + 1 < KT)   // prefetch next W2 K-slab
        __builtin_prefetch(wBase + (size_t)(KT + kt + 1)*NT*32, 0, 3);
      #pragma unroll
      for (int n = 0; n < 4; ++n) {
        v16bf b = wBase[((size_t)(KT + kt)*NT + n)*32];
        acc[n] = __builtin_amdgcn_wmma_f32_16x16x32_bf16(
                     false, a, false, b, (short)0, acc[n], false, false);
      }
    }
    #pragma unroll
    for (int n = 0; n < 4; ++n) {
      int col = (wave*4 + n)*16 + l16;
      int ac  = col & (H_ - 1);            // alpha tiled x4 over gate blocks
      #pragma unroll
      for (int r = 0; r < 8; ++r) {
        int row = r + 8*lg;
        float g1 = sG[row*G_ + col];
        float a  = sAlpha[row*H_ + ac];
        sG[row*G_ + col] = fmaf(a, g1 - acc[n][r], acc[n][r]);
      }
    }
    __syncthreads();

    // -------- Phase 4: LSTM cell update, h -> A frags, output projection ---
    float partial = 0.0f;
    #pragma unroll
    for (int jj = 0; jj < 8; ++jj) {
      int j = uJ0 + jj;
      float gi = sG[uRow*G_ + j];
      float gf = sG[uRow*G_ +   H_ + j];
      float gg = sG[uRow*G_ + 2*H_ + j];
      float go = sG[uRow*G_ + 3*H_ + j];
      c[jj] = fast_sig(gf)*c[jj] + fast_sig(gi)*fast_tanh(gg);
      float h = fast_sig(go)*fast_tanh(c[jj]);
      // Scatter h into next step's A-fragment layout.
      int kt2 = j >> 5;
      int k   = j & 31;
      int flg = (k >> 3) & 1;
      int fe  = (k & 7) + 8*(k >> 4);
      sAB[kt2*512 + (uRow + 16*flg)*16 + fe] = (bf16_t)h;
      partial = fmaf(h, sLinW[j], partial);
    }
    // out[b,t] = h . lin_W + lin_b  (full-wave tree reduction, 32 lanes/row)
    partial += __shfl_xor(partial, 16, 32);
    partial += __shfl_xor(partial,  8, 32);
    partial += __shfl_xor(partial,  4, 32);
    partial += __shfl_xor(partial,  2, 32);
    partial += __shfl_xor(partial,  1, 32);
    if (lane == 0)
      out[outOff + (unsigned)t] = partial + lb;
    __syncthreads();
  }
}

// ---------------------------------------------------------------------------
extern "C" void kernel_launch(void* const* d_in, const int* in_sizes, int n_in,
                              void* d_out, int out_size, void* d_ws, size_t ws_size,
                              hipStream_t stream) {
  const int*   e   = (const int*)  d_in[0];
  const float* x   = (const float*)d_in[1];
  const float* emb = (const float*)d_in[2];
  const float* Wx1 = (const float*)d_in[3];
  const float* Wx2 = (const float*)d_in[4];
  const float* Wh1 = (const float*)d_in[5];
  const float* Wh2 = (const float*)d_in[6];
  const float* b1  = (const float*)d_in[7];
  const float* b2  = (const float*)d_in[8];
  const float* We  = (const float*)d_in[9];
  const float* be  = (const float*)d_in[10];
  const float* lW  = (const float*)d_in[11];
  const float* lb  = (const float*)d_in[12];
  float* out = (float*)d_out;

  bf16_t* wFrags  = (bf16_t*)d_ws;
  bf16_t* weFrags = wFrags + WFRAG_ELEMS;

  int totalPrep = WFRAG_ELEMS + WEFRAG_ELEMS;
  prep_weights<<<(totalPrep + 255)/256, 256, 0, stream>>>(
      Wx1, Wx2, Wh1, Wh2, b1, b2, We, wFrags, weFrags);
  lstm_persistent<<<B_/ROWS, THREADS, 0, stream>>>(
      e, x, emb, be, lW, lb, wFrags, weFrags, out);
}
